// GCN_31593779430135
// MI455X (gfx1250) — matrix operand order repaired and checked by reading, not compiled, not run on hardware
//
#include <hip/hip_runtime.h>

typedef __attribute__((ext_vector_type(16))) __bf16 v16bf;
typedef __attribute__((ext_vector_type(8)))  __bf16 v8bf;
typedef __attribute__((ext_vector_type(8)))  float  v8f;

#define N_NODES   50000
#define N_EDGES   800000
#define IN_FEATS  128
#define N_HIDDEN  64
#define N_CLASSES 40

// ---------------------------------------------------------------------------
// Utility: fill a float buffer with a constant (used to zero accumulators).
// ---------------------------------------------------------------------------
__global__ void gcn_fill(float* __restrict__ p, int n, float v) {
    for (int i = blockIdx.x * blockDim.x + threadIdx.x; i < n;
         i += gridDim.x * blockDim.x)
        p[i] = v;
}

// ---------------------------------------------------------------------------
// Prepass: convert W[K, Fout] (f32 row-major) -> Wt[Fout, K] (bf16, each
// output column contiguous). Runs once per layer; Wt stays L2-resident.
// ---------------------------------------------------------------------------
__global__ void gcn_wt_prep(const float* __restrict__ W, __bf16* __restrict__ Wt,
                            int K, int Fout) {
    int total = K * Fout;
    for (int i = blockIdx.x * blockDim.x + threadIdx.x; i < total;
         i += gridDim.x * blockDim.x) {
        int k = i / Fout;
        int n = i % Fout;
        Wt[n * K + k] = (__bf16)W[i];
    }
}

// ---------------------------------------------------------------------------
// Degree count over E real edges + N self loops (atomic f32 adds).
// ---------------------------------------------------------------------------
__global__ void gcn_degree(const int* __restrict__ dst, float* __restrict__ deg,
                           int E, int N) {
    int total = E + N;
    for (int i = blockIdx.x * blockDim.x + threadIdx.x; i < total;
         i += gridDim.x * blockDim.x) {
        int d = (i < E) ? dst[i] : (i - E);
        atomicAdd(&deg[d], 1.0f);
    }
}

__global__ void gcn_dinv(float* __restrict__ deg, int N) {
    for (int i = blockIdx.x * blockDim.x + threadIdx.x; i < N;
         i += gridDim.x * blockDim.x)
        deg[i] = rsqrtf(fmaxf(deg[i], 1.0f));
}

// ---------------------------------------------------------------------------
// GEMM: out[N, Fout] = A[N, K] @ W[K, Fout] via v_wmma_f32_16x16x32_bf16.
//
// Block = 128 threads = 4 waves; block owns a 16-row slab (N = 3125 * 16);
// wave w owns column tile [16w, 16w+16).
//
// A tile staged once per block into padded bf16 LDS (rows +8 bf16 so the
// stride-K fragment reads hit distinct banks). B fragments come straight
// from the transposed bf16 Wt in global memory: two runs of 8 contiguous K-values per
// lane = two global_load_b128, L0/L2-resident across all blocks.
// ---------------------------------------------------------------------------
template <int K, int Fout>
__global__ void __launch_bounds__(128)
gcn_gemm_wmma(const float* __restrict__ A, const __bf16* __restrict__ Wt,
              float* __restrict__ out) {
    constexpr int KP = K + 8;                 // padded LDS row stride (bf16)
    __shared__ __bf16 sA[16 * KP];

    const int tid     = threadIdx.x;
    const int rowBase = blockIdx.x * 16;

    // ---- stage A tile (16 x K) -> bf16 LDS, coalesced global reads ----
    #pragma unroll
    for (int i = tid; i < 16 * K; i += 128) {
        const int r = i / K;                  // K is a power of two -> shifts
        const int k = i % K;
        sA[r * KP + k] = (__bf16)A[(rowBase + r) * K + k];
    }
    __syncthreads();

    const int lane = tid & 31;
    const int wave = tid >> 5;
    const int half = lane >> 4;               // which 16-lane half of the wave
    const int m16  = lane & 15;

    const int n0 = wave * 16;
    if (n0 >= Fout) return;                   // wave-uniform (layer 3: 3 tiles)

    const int  col   = n0 + m16;              // B/D column held by this lane
    const bool colOK = (col < Fout);
    const int  ccol  = colOK ? col : (Fout - 1);  // clamp: loads in-bounds;
                                              // junk columns never stored.

    const __bf16* __restrict__ aRow = sA + m16 * KP;
    const __bf16* __restrict__ wCol = Wt + ccol * K;   // contiguous column

    v8f acc = {};
    #pragma unroll
    for (int k0 = 0; k0 < K; k0 += 32) {
        // A fragment: 16x32 bf16 ISA layout.
        // half 0: VGPR0-3 = K 0..7,  VGPR4-7 = K 16..23
        // half 1: VGPR0-3 = K 8..15, VGPR4-7 = K 24..31
        const int ka = k0 + half * 8;
        union { v16bf v; v8bf h[2]; } a;
        a.h[0] = *(const v8bf*)(aRow + ka);        // ds_load_b128
        a.h[1] = *(const v8bf*)(aRow + ka + 16);   // ds_load_b128

        // B fragment: 32x16 bf16; lane holds column ccol, same K layout as A:
        // K k0+8*half..+7 and k0+16+8*half..+7.
        const int kb = k0 + half * 8;
        union { v16bf v; v8bf h[2]; } b;
        b.h[0] = *(const v8bf*)(wCol + kb);        // global_load_b128
        b.h[1] = *(const v8bf*)(wCol + kb + 16);   // global_load_b128

        acc = __builtin_amdgcn_wmma_f32_16x16x32_bf16(
            /*neg_a=*/false, a.v, /*neg_b=*/false, b.v,
            /*c_mod=*/(short)0, acc, /*reuse_a=*/false, /*reuse_b=*/false);
    }

    if (colOK) {
        // D layout: VGPR r -> M = r + 8*half, N = lane % 16
        #pragma unroll
        for (int r = 0; r < 8; ++r) {
            const int orow = rowBase + r + half * 8;
            out[orow * Fout + col] = acc[r];
        }
    }
}

// ---------------------------------------------------------------------------
// Edge aggregation: agg[dst] += t[src] * dinv[src]*dinv[dst] over E real
// edges + N self loops. One wave per edge, lanes stride features (coalesced
// 128B bursts), global f32 atomics into destination rows.
// ---------------------------------------------------------------------------
__global__ void gcn_edge_agg(const float* __restrict__ t,
                             const int* __restrict__ src,
                             const int* __restrict__ dst,
                             const float* __restrict__ dinv,
                             float* __restrict__ agg,
                             int E, int N, int F) {
    const int lane  = threadIdx.x & 31;
    const int wid   = (blockIdx.x * blockDim.x + threadIdx.x) >> 5;
    const int nw    = (gridDim.x * blockDim.x) >> 5;
    const int total = E + N;

    for (int e = wid; e < total; e += nw) {
        const int s = (e < E) ? src[e] : (e - E);
        const int d = (e < E) ? dst[e] : (e - E);
        const float norm = dinv[s] * dinv[d];
        const float* __restrict__ ts = t + s * F;
        float* __restrict__ ad = agg + d * F;
        for (int f = lane; f < F; f += 32)
            atomicAdd(&ad[f], ts[f] * norm);
    }
}

// ---------------------------------------------------------------------------
// out = agg + bias, optional ReLU.
// ---------------------------------------------------------------------------
__global__ void gcn_bias_act(const float* __restrict__ agg,
                             const float* __restrict__ bias,
                             float* __restrict__ out,
                             int total, int F, int doRelu) {
    for (int i = blockIdx.x * blockDim.x + threadIdx.x; i < total;
         i += gridDim.x * blockDim.x) {
        float v = agg[i] + bias[i % F];
        out[i] = doRelu ? fmaxf(v, 0.0f) : v;
    }
}

// ---------------------------------------------------------------------------
extern "C" void kernel_launch(void* const* d_in, const int* in_sizes, int n_in,
                              void* d_out, int out_size, void* d_ws, size_t ws_size,
                              hipStream_t stream) {
    (void)in_sizes; (void)n_in; (void)out_size; (void)ws_size;

    const float* x  = (const float*)d_in[0];
    const int*   ei = (const int*)d_in[1];
    const float* W1 = (const float*)d_in[2];
    const float* b1 = (const float*)d_in[3];
    const float* W2 = (const float*)d_in[4];
    const float* b2 = (const float*)d_in[5];
    const float* W3 = (const float*)d_in[6];
    const float* b3 = (const float*)d_in[7];

    const int* src = ei;            // edge_index[0]
    const int* dst = ei + N_EDGES;  // edge_index[1]

    float* ws   = (float*)d_ws;
    float* dinv = ws;                        // N floats
    float* tbuf = dinv + N_NODES;            // N * 64 (GEMM output)
    float* agg  = tbuf + N_NODES * 64;       // N * 64 (aggregation)
    float* hbuf = agg  + N_NODES * 64;       // N * 64 (layer activations)
    // bf16 transposed weights (16B-aligned: offset is a multiple of 16 bytes)
    __bf16* Wt1 = (__bf16*)(hbuf + N_NODES * 64);          // 64 * 128
    __bf16* Wt2 = Wt1 + N_HIDDEN * IN_FEATS;               // 64 * 64
    __bf16* Wt3 = Wt2 + N_HIDDEN * N_HIDDEN;               // 40 * 64
    float* out  = (float*)d_out;             // N * 40

    const dim3 blk(256);
    const dim3 gridFill(1024);
    const dim3 gridEdge(2048);
    const dim3 gridW(32);
    const dim3 gridGemm(N_NODES / 16);       // 3125 slabs of 16 rows
    const dim3 blkGemm(128);                 // 4 waves

    // --- weight prepass (f32 row-major -> bf16 column-major) ---
    gcn_wt_prep<<<gridW, blk, 0, stream>>>(W1, Wt1, IN_FEATS, N_HIDDEN);
    gcn_wt_prep<<<gridW, blk, 0, stream>>>(W2, Wt2, N_HIDDEN, N_HIDDEN);
    gcn_wt_prep<<<gridW, blk, 0, stream>>>(W3, Wt3, N_HIDDEN, N_CLASSES);

    // --- normalization coefficients ---
    gcn_fill<<<gridFill, blk, 0, stream>>>(dinv, N_NODES, 0.0f);
    gcn_degree<<<gridEdge, blk, 0, stream>>>(dst, dinv, N_EDGES, N_NODES);
    gcn_dinv<<<gridFill, blk, 0, stream>>>(dinv, N_NODES);

    // --- layer 1: x[N,128] @ W1[128,64] -> agg -> relu(hbuf) ---
    gcn_gemm_wmma<IN_FEATS, N_HIDDEN><<<gridGemm, blkGemm, 0, stream>>>(x, Wt1, tbuf);
    gcn_fill<<<gridFill, blk, 0, stream>>>(agg, N_NODES * N_HIDDEN, 0.0f);
    gcn_edge_agg<<<gridEdge, blk, 0, stream>>>(tbuf, src, dst, dinv, agg,
                                               N_EDGES, N_NODES, N_HIDDEN);
    gcn_bias_act<<<gridEdge, blk, 0, stream>>>(agg, b1, hbuf,
                                               N_NODES * N_HIDDEN, N_HIDDEN, 1);

    // --- layer 2: hbuf[N,64] @ W2[64,64] -> agg -> relu(hbuf) ---
    gcn_gemm_wmma<N_HIDDEN, N_HIDDEN><<<gridGemm, blkGemm, 0, stream>>>(hbuf, Wt2, tbuf);
    gcn_fill<<<gridFill, blk, 0, stream>>>(agg, N_NODES * N_HIDDEN, 0.0f);
    gcn_edge_agg<<<gridEdge, blk, 0, stream>>>(tbuf, src, dst, dinv, agg,
                                               N_EDGES, N_NODES, N_HIDDEN);
    gcn_bias_act<<<gridEdge, blk, 0, stream>>>(agg, b2, hbuf,
                                               N_NODES * N_HIDDEN, N_HIDDEN, 1);

    // --- layer 3: hbuf[N,64] @ W3[64,40] -> agg -> out (no relu) ---
    gcn_gemm_wmma<N_HIDDEN, N_CLASSES><<<gridGemm, blkGemm, 0, stream>>>(hbuf, Wt3, tbuf);
    gcn_fill<<<gridFill, blk, 0, stream>>>(agg, N_NODES * N_CLASSES, 0.0f);
    gcn_edge_agg<<<gridEdge, blk, 0, stream>>>(tbuf, src, dst, dinv, agg,
                                               N_EDGES, N_NODES, N_CLASSES);
    gcn_bias_act<<<gridEdge, blk, 0, stream>>>(agg, b3, out,
                                               N_NODES * N_CLASSES, N_CLASSES, 0);
}
